// mymodel_71640054497431
// MI455X (gfx1250) — compile-verified, matrix operand order checked by reading, not run on hardware
//
#include <hip/hip_runtime.h>
#include <cstdint>

// ---------------------------------------------------------------------------
// MI455X (gfx1250) Slot-Attention forward.
// Heavy GEMMs: v_wmma_f32_16x16x32_bf16, LDS tiles staged by the Tensor Data
// Mover (tensor_load_to_lds, TENSORcnt double-buffered).
// Input order assumes jax tree_flatten (sorted dict keys):
//   0: aud (2048,16,512)   1: img (2048,49,512)
//   2..21  : aud params sorted, 22..41 : img params, 42 : slots (1,8,512)
// Output (f32, concat): img_slots, aud_slots, img_q, aud_q,
//                       ii_attn(b,8,49), ia_attn(b,8,16), aa_attn(b,8,16),
//                       ai_attn(b,8,49)
// ---------------------------------------------------------------------------

typedef __bf16 bf16_t;
typedef bf16_t bf16x8  __attribute__((ext_vector_type(8)));
typedef bf16_t bf16x16 __attribute__((ext_vector_type(16)));
typedef float  f32x8   __attribute__((ext_vector_type(8)));
typedef unsigned int u32x4 __attribute__((ext_vector_type(4)));
typedef int          i32x4 __attribute__((ext_vector_type(4)));
typedef int          i32x8 __attribute__((ext_vector_type(8)));

static constexpr int BATCH = 2048;
static constexpr int NI    = 49;
static constexpr int NA    = 16;
static constexpr int DIM   = 512;
static constexpr int NSLOT = 8;
static constexpr int HF    = 1024;
static constexpr int MSLOT = BATCH * NSLOT;   // 16384
static constexpr int MIMG  = BATCH * NI;      // 100352
static constexpr int MAUD  = BATCH * NA;      // 32768

// ------------------------------ device helpers ----------------------------

__device__ __forceinline__ f32x8 wmma_bf16(bf16x16 a, bf16x16 b, f32x8 c) {
#if defined(__HIP_DEVICE_COMPILE__)
  return __builtin_amdgcn_wmma_f32_16x16x32_bf16(false, a, false, b, (short)0, c,
                                                 false, false);
#else
  return c;
#endif
}

// Load a 16x32 bf16 fragment given a per-lane base (row = lane&15,
// K chunk base = (lane>>4)*8, second chunk at +16 elements).
__device__ __forceinline__ bf16x16 frag_ld(const bf16_t* p) {
  bf16x8 lo = *(const bf16x8*)(p);
  bf16x8 hi = *(const bf16x8*)(p + 16);
  return __builtin_shufflevector(lo, hi, 0, 1, 2, 3, 4, 5, 6, 7, 8, 9, 10, 11,
                                 12, 13, 14, 15);
}

// --------------------- Tensor Data Mover tile load ------------------------
// 2D tile: rows x 32 bf16 elements, row stride = strideElems (bf16),
// DMA'd global -> LDS with hardware padding of 4 DWORDs every 16 DWORDs
// (i.e. 64B data + 16B pad = 80B LDS row = LDT=40 bf16 stride).
static constexpr int LDT = 40;  // padded LDS row stride (bf16 elems)

__device__ __forceinline__ void tdm_load_tile(uint32_t ldsByte,
                                              const bf16_t* gptr,
                                              int strideElems, int rows) {
#if defined(__HIP_DEVICE_COMPILE__)
  const uint64_t ga = (uint64_t)(uintptr_t)gptr;
  u32x4 g0;
  g0[0] = 1u;                               // count=1, user mode, no gather
  g0[1] = ldsByte;                          // lds_addr
  g0[2] = (uint32_t)ga;                     // global_addr[31:0]
  g0[3] = (uint32_t)((ga >> 32) & 0x1FFFFFFu) | (2u << 30);  // ga[56:32], type=2
  i32x8 g1;
  // word0: workgroup_mask=0 | data_size=1(2B)<<16 | pad_enable<<20 |
  //        pad_interval=3(16 DW)<<22 | pad_amount=3(4 DW)<<25
  g1[0] = (1 << 16) | (1 << 20) | (3 << 22) | (3 << 25);
  const uint32_t td0 = (uint32_t)strideElems;   // tensor_dim0 (>= tile_dim0)
  const uint32_t td1 = 1u << 20;                // tensor_dim1 (large, no OOB)
  g1[1] = (int)((td0 & 0xFFFFu) << 16);               // abar=0 | td0 lo16
  g1[2] = (int)((td0 >> 16) | ((td1 & 0xFFFFu) << 16));  // td0 hi | td1 lo
  g1[3] = (int)((td1 >> 16) | (32u << 16));           // td1 hi | tile_dim0=32
  g1[4] = (int)((uint32_t)rows);                      // tile_dim1 | tile_dim2=0
  g1[5] = (int)(uint32_t)strideElems;                 // dim0_stride lo32
  g1[6] = 0;                                          // dim0_stride hi | d1s lo
  g1[7] = 0;
  const i32x4 gz = {0, 0, 0, 0};
#if __clang_major__ >= 23
  const i32x8 gz8 = {0, 0, 0, 0, 0, 0, 0, 0};
  __builtin_amdgcn_tensor_load_to_lds(g0, g1, gz, gz, gz8, 0);
#else
  __builtin_amdgcn_tensor_load_to_lds(g0, g1, gz, gz, 0);
#endif
#else
  (void)ldsByte; (void)gptr; (void)strideElems; (void)rows;
#endif
}

__device__ __forceinline__ uint32_t lds_off(const void* p) {
  return (uint32_t)(uintptr_t)p;  // generic LDS address: low 32 bits = offset
}

// ------------------------------ GEMM kernel -------------------------------
// C[M,N] = A[M,K](bf16) x W[N,K]^T(bf16) + bias[N]
// optional relu, residual add (f32), f32 and/or bf16 outputs.
// M % 128 == 0, N % 128 == 0, K % 32 == 0 (all shapes here satisfy this).
// Block 256 threads = 8 waves (4x2); wave tile 32x64; block tile 128x128.
// A/B K-tiles DMA'd by the TDM, double-buffered on TENSORcnt.
__global__ __launch_bounds__(256) void gemm_bias(
    const bf16_t* __restrict__ A, const bf16_t* __restrict__ W,
    const float* __restrict__ bias, const float* __restrict__ residual,
    float* __restrict__ outF, bf16_t* __restrict__ outB,
    int M, int N, int K, int relu) {
  __shared__ bf16_t As[2][128 * LDT];
  __shared__ bf16_t Bs[2][128 * LDT];

  const int mBase = blockIdx.y * 128;
  const int nBase = blockIdx.x * 128;
  const int t = threadIdx.x, wave = t >> 5, lane = t & 31;
  const int wm = wave >> 1, wn = wave & 1;

  f32x8 acc[2][4] = {};

  const bf16_t* Ab = A + (size_t)mBase * K;
  const bf16_t* Wb = W + (size_t)nBase * K;
  const int nk = K / 32;

  if (wave == 0) {
    tdm_load_tile(lds_off(&As[0][0]), Ab, K, 128);
    tdm_load_tile(lds_off(&Bs[0][0]), Wb, K, 128);
  }
  for (int i = 0; i < nk; ++i) {
    if (wave == 0) {
      if (i + 1 < nk) {
        tdm_load_tile(lds_off(&As[(i + 1) & 1][0]), Ab + (i + 1) * 32, K, 128);
        tdm_load_tile(lds_off(&Bs[(i + 1) & 1][0]), Wb + (i + 1) * 32, K, 128);
        __builtin_amdgcn_s_wait_tensorcnt(2);  // tiles of step i resident
      } else {
        __builtin_amdgcn_s_wait_tensorcnt(0);
      }
    }
    __syncthreads();
    const bf16_t* as = As[i & 1];
    const bf16_t* bs = Bs[i & 1];
    const int fr = lane & 15;
    const int fc = (lane >> 4) << 3;
    bf16x16 afrag[2];
#pragma unroll
    for (int mi = 0; mi < 2; ++mi)
      afrag[mi] = frag_ld(&as[(wm * 32 + mi * 16 + fr) * LDT + fc]);
#pragma unroll
    for (int j = 0; j < 4; ++j) {
      bf16x16 bfr = frag_ld(&bs[(wn * 64 + j * 16 + fr) * LDT + fc]);
#pragma unroll
      for (int mi = 0; mi < 2; ++mi)
        acc[mi][j] = wmma_bf16(afrag[mi], bfr, acc[mi][j]);
    }
    __syncthreads();  // all waves done with buf (i&1) before its re-fill
  }

  // epilogue: lane holds n = lane&15; rows m = (lane>>4)*8 + i
  const int n0 = nBase + wn * 64 + (lane & 15);
  const int m0 = mBase + wm * 32 + ((lane >> 4) << 3);
#pragma unroll
  for (int mi = 0; mi < 2; ++mi) {
#pragma unroll
    for (int j = 0; j < 4; ++j) {
      const int n = n0 + j * 16;
      const float bv = bias ? bias[n] : 0.0f;
#pragma unroll
      for (int i = 0; i < 8; ++i) {
        const size_t idx = (size_t)(m0 + mi * 16 + i) * N + n;
        float v = acc[mi][j][i] + bv;
        if (relu) v = v > 0.0f ? v : 0.0f;
        if (residual) v += residual[idx];
        if (outF) outF[idx] = v;
        if (outB) outB[idx] = (bf16_t)v;
      }
    }
  }
}

// ------------------------------ LayerNorm ---------------------------------
__global__ __launch_bounds__(256) void ln_rows(
    const float* __restrict__ x, const float* __restrict__ w,
    const float* __restrict__ b, bf16_t* __restrict__ out, int rows) {
  const int wave = threadIdx.x >> 5, lane = threadIdx.x & 31;
  const int row = blockIdx.x * 8 + wave;
  if (row >= rows) return;
  const float* xr = x + (size_t)row * DIM;
  float v[16], s = 0.0f;
#pragma unroll
  for (int i = 0; i < 16; ++i) { v[i] = xr[lane + i * 32]; s += v[i]; }
#pragma unroll
  for (int o = 16; o; o >>= 1) s += __shfl_xor(s, o, 32);
  const float mean = s * (1.0f / DIM);
  float q = 0.0f;
#pragma unroll
  for (int i = 0; i < 16; ++i) { float d = v[i] - mean; q += d * d; }
#pragma unroll
  for (int o = 16; o; o >>= 1) q += __shfl_xor(q, o, 32);
  const float rstd = rsqrtf(q * (1.0f / DIM) + 1e-5f);
  bf16_t* orow = out + (size_t)row * DIM;
#pragma unroll
  for (int i = 0; i < 16; ++i) {
    const int c = lane + i * 32;
    orow[c] = (bf16_t)((v[i] - mean) * rstd * w[c] + b[c]);
  }
}

// --------------------------- slot attention -------------------------------
static constexpr int ATS = 68;  // LDS stride for 16x64 attention tile

__global__ __launch_bounds__(128) void slot_attention(
    const bf16_t* __restrict__ q,    // (B*8, 512)
    const bf16_t* __restrict__ k,    // (B*Nk, 512)
    const bf16_t* __restrict__ v,    // (B*Nk, 512) or null
    float* __restrict__ attn_out,    // (B, 8, Nk) or null
    bf16_t* __restrict__ upd_out,    // (B*8, 512) or null
    int Nk, float scale) {
  __shared__ float att[16 * ATS];
  __shared__ float rsum[8];
  const int bi = blockIdx.x;
  const int t = threadIdx.x, wave = t >> 5, lane = t & 31;
  const int NT = (Nk + 15) >> 4;

  for (int i = t; i < 16 * ATS; i += 128) att[i] = 0.0f;
  __syncthreads();

  // dots = q @ k^T : each wave one 16(slot-padded)x16(token) tile
  if (wave < NT) {
    f32x8 acc = {};
    const int srow = bi * NSLOT + ((lane & 15) & 7);  // dup slot rows 8..15
    int tok = wave * 16 + (lane & 15);
    if (tok >= Nk) tok = 0;  // masked later
    const int krow = bi * Nk + tok;
    const int fc = (lane >> 4) << 3;
    const bf16_t* qp = q + (size_t)srow * DIM + fc;
    const bf16_t* kp = k + (size_t)krow * DIM + fc;
    for (int k0 = 0; k0 < DIM; k0 += 32)
      acc = wmma_bf16(frag_ld(qp + k0), frag_ld(kp + k0), acc);
    if (lane < 16) {
#pragma unroll
      for (int i = 0; i < 8; ++i)
        att[i * ATS + wave * 16 + lane] = acc[i] * scale;
    }
  }
  __syncthreads();

  // softmax over the 8 slots per token, + eps
  if (t < 64) {
    const int j = t;
    if (j < Nk) {
      float c8[8], m = -1e30f;
#pragma unroll
      for (int i = 0; i < 8; ++i) { c8[i] = att[i * ATS + j]; m = fmaxf(m, c8[i]); }
      float s = 0.0f;
#pragma unroll
      for (int i = 0; i < 8; ++i) { c8[i] = __expf(c8[i] - m); s += c8[i]; }
      const float inv = 1.0f / s;
#pragma unroll
      for (int i = 0; i < 8; ++i) att[i * ATS + j] = c8[i] * inv + 1e-8f;
    } else {
#pragma unroll
      for (int i = 0; i < 8; ++i) att[i * ATS + j] = 0.0f;
    }
  }
  __syncthreads();

  // renormalize over tokens
  if (t < 8) {
    float s = 0.0f;
    for (int j = 0; j < Nk; ++j) s += att[t * ATS + j];
    rsum[t] = 1.0f / s;
  }
  __syncthreads();
  if (t < 64 && t < Nk) {
#pragma unroll
    for (int i = 0; i < 8; ++i) {
      const float a = att[i * ATS + t] * rsum[i];
      att[i * ATS + t] = a;
      if (attn_out) attn_out[((size_t)bi * 8 + i) * Nk + t] = a;
    }
  }
  __syncthreads();

  // upd = attn @ v (A from LDS, B = v columns)
  if (v) {
    const int KT = (NT * 16 + 31) & ~31;  // 64 (img) or 32 (aud)
    const int fc = (lane >> 4) << 3;
    for (int nt = wave; nt < DIM / 16; nt += 4) {
      f32x8 acc = {};
      const int dcol = nt * 16 + (lane & 15);
      for (int k0 = 0; k0 < KT; k0 += 32) {
        bf16x16 af, bfr;
        const int r = lane & 15;  // att rows 8..15 are zero
#pragma unroll
        for (int i = 0; i < 8; ++i) af[i] = (bf16_t)att[r * ATS + k0 + fc + i];
#pragma unroll
        for (int i = 0; i < 8; ++i)
          af[8 + i] = (bf16_t)att[r * ATS + k0 + fc + 16 + i];
#pragma unroll
        for (int i = 0; i < 8; ++i) {
          int kk = k0 + fc + i;      if (kk >= Nk) kk = 0;
          bfr[i] = v[((size_t)bi * Nk + kk) * DIM + dcol];
        }
#pragma unroll
        for (int i = 0; i < 8; ++i) {
          int kk = k0 + fc + 16 + i; if (kk >= Nk) kk = 0;
          bfr[8 + i] = v[((size_t)bi * Nk + kk) * DIM + dcol];
        }
        acc = wmma_bf16(af, bfr, acc);
      }
      if (lane < 16) {
#pragma unroll
        for (int i = 0; i < 8; ++i)
          upd_out[((size_t)bi * 8 + i) * DIM + dcol] = (bf16_t)acc[i];
      }
    }
  }
}

// ------------------------------ GRU gates ---------------------------------
__global__ __launch_bounds__(256) void gru_ew(const float* __restrict__ gi,
                                              const float* __restrict__ gh,
                                              float* __restrict__ slots) {
  const size_t idx = (size_t)blockIdx.x * blockDim.x + threadIdx.x;
  const size_t row = idx >> 9;
  const int d = (int)(idx & 511);
  const size_t base = row * (3 * DIM);
  const float r = 1.0f / (1.0f + __expf(-(gi[base + d] + gh[base + d])));
  const float z =
      1.0f / (1.0f + __expf(-(gi[base + DIM + d] + gh[base + DIM + d])));
  const float n = tanhf(gi[base + 2 * DIM + d] + r * gh[base + 2 * DIM + d]);
  const float h = slots[idx];
  slots[idx] = (1.0f - z) * n + z * h;
}

// --------------------------- small utilities ------------------------------
__global__ __launch_bounds__(256) void slots_init(const float* __restrict__ sp,
                                                  float* __restrict__ sF,
                                                  bf16_t* __restrict__ sB) {
  const size_t idx = (size_t)blockIdx.x * blockDim.x + threadIdx.x;
  const int d = (int)(idx & 511);
  const int slot = (int)((idx >> 9) & 7);
  const float v = sp[slot * DIM + d];
  sF[idx] = v;
  sB[idx] = (bf16_t)v;
}

__global__ __launch_bounds__(256) void cvt_bf16(const float* __restrict__ x,
                                                bf16_t* __restrict__ y, int n) {
  const int i = blockIdx.x * blockDim.x + threadIdx.x;
  if (i < n) y[i] = (bf16_t)x[i];
}

// ------------------------------ host side ---------------------------------
extern "C" void kernel_launch(void* const* d_in, const int* in_sizes, int n_in,
                              void* d_out, int out_size, void* d_ws,
                              size_t ws_size, hipStream_t stream) {
  (void)in_sizes; (void)n_in; (void)out_size; (void)ws_size;

  const float* aud = (const float*)d_in[0];
  const float* img = (const float*)d_in[1];
  const float* Pa[20]; const float* Pi[20];
  for (int i = 0; i < 20; ++i) {
    Pa[i] = (const float*)d_in[2 + i];
    Pi[i] = (const float*)d_in[22 + i];
  }
  // per-modality sorted param index map:
  // 0 gru_bhh  1 gru_bih  2 gru_whh  3 gru_wih  4 k_b  5 k_w
  // 6 ln_ff_b  7 ln_ff_w  8 ln_in_b  9 ln_in_w 10 ln_slots_b 11 ln_slots_w
  // 12 mlp_b1 13 mlp_b2 14 mlp_w1 15 mlp_w2 16 q_b 17 q_w 18 v_b 19 v_w
  const float* slotsParam = (const float*)d_in[42];

  // ---- workspace arena
  char* w = (char*)d_ws;
  size_t off = 0;
  auto alloc = [&](size_t bytes) {
    void* p = w + off;
    off = (off + bytes + 255) & ~(size_t)255;
    return p;
  };
  auto allocB = [&](size_t elems) { return (bf16_t*)alloc(elems * 2); };
  auto allocF = [&](size_t elems) { return (float*)alloc(elems * 4); };

  bf16_t *QwI = allocB(DIM * DIM), *KwI = allocB(DIM * DIM),
         *VwI = allocB(DIM * DIM), *WihI = allocB(3 * DIM * DIM),
         *WhhI = allocB(3 * DIM * DIM), *W1I = allocB(HF * DIM),
         *W2I = allocB(DIM * HF);
  bf16_t *QwA = allocB(DIM * DIM), *KwA = allocB(DIM * DIM),
         *VwA = allocB(DIM * DIM), *WihA = allocB(3 * DIM * DIM),
         *WhhA = allocB(3 * DIM * DIM), *W1A = allocB(HF * DIM),
         *W2A = allocB(DIM * HF);

  // imgN region is reused for gi (f32) after projections; audN region for gh.
  const size_t giBytes = (size_t)MSLOT * 3 * DIM * 4;  // 100.7 MB
  char* regI = (char*)alloc((size_t)MIMG * DIM * 2 > giBytes
                                ? (size_t)MIMG * DIM * 2 : giBytes);
  char* regA = (char*)alloc((size_t)MAUD * DIM * 2 > giBytes
                                ? (size_t)MAUD * DIM * 2 : giBytes);
  bf16_t* imgN = (bf16_t*)regI;  float* giF = (float*)regI;
  bf16_t* audN = (bf16_t*)regA;  float* ghF = (float*)regA;

  bf16_t* imgK = allocB((size_t)MIMG * DIM);
  bf16_t* imgV = allocB((size_t)MIMG * DIM);
  bf16_t* audK = allocB((size_t)MAUD * DIM);
  bf16_t* audV = allocB((size_t)MAUD * DIM);
  float*  slotsF = allocF((size_t)MSLOT * DIM);
  bf16_t* prevB  = allocB((size_t)MSLOT * DIM);
  bf16_t* qB     = allocB((size_t)MSLOT * DIM);
  bf16_t* updB   = allocB((size_t)MSLOT * DIM);
  bf16_t* lnS    = allocB((size_t)MSLOT * DIM);
  bf16_t* h1B    = allocB((size_t)MSLOT * HF);

  // ---- output regions
  float* out = (float*)d_out;
  float* oImgSlots = out;
  float* oAudSlots = out + (size_t)MSLOT * DIM;
  float* oImgQ     = out + 2 * (size_t)MSLOT * DIM;
  float* oAudQ     = out + 3 * (size_t)MSLOT * DIM;
  float* oII = out + 4 * (size_t)MSLOT * DIM;          // (b,8,49)
  float* oIA = oII + (size_t)BATCH * NSLOT * NI;       // (b,8,16)
  float* oAA = oIA + (size_t)BATCH * NSLOT * NA;       // (b,8,16)
  float* oAI = oAA + (size_t)BATCH * NSLOT * NA;       // (b,8,49)

  auto cvt = [&](const float* src, bf16_t* dst, int n) {
    cvt_bf16<<<(n + 255) / 256, 256, 0, stream>>>(src, dst, n);
  };
  auto gemm = [&](const bf16_t* A, const bf16_t* W_, const float* bias,
                  const float* resid, float* oF, bf16_t* oB, int M, int N,
                  int K, int relu) {
    dim3 g(N / 128, M / 128);
    gemm_bias<<<g, 256, 0, stream>>>(A, W_, bias, resid, oF, oB, M, N, K, relu);
  };

  // ---- convert weights to bf16
  cvt(Pi[17], QwI, DIM * DIM);  cvt(Pi[5], KwI, DIM * DIM);
  cvt(Pi[19], VwI, DIM * DIM);  cvt(Pi[3], WihI, 3 * DIM * DIM);
  cvt(Pi[2], WhhI, 3 * DIM * DIM);
  cvt(Pi[14], W1I, HF * DIM);   cvt(Pi[15], W2I, DIM * HF);
  cvt(Pa[17], QwA, DIM * DIM);  cvt(Pa[5], KwA, DIM * DIM);
  cvt(Pa[19], VwA, DIM * DIM);  cvt(Pa[3], WihA, 3 * DIM * DIM);
  cvt(Pa[2], WhhA, 3 * DIM * DIM);
  cvt(Pa[14], W1A, HF * DIM);   cvt(Pa[15], W2A, DIM * HF);

  // ---- input LayerNorm + k/v projections (imgN/audN then recycled gi/gh)
  ln_rows<<<MIMG / 8, 256, 0, stream>>>(img, Pi[9], Pi[8], imgN, MIMG);
  ln_rows<<<MAUD / 8, 256, 0, stream>>>(aud, Pa[9], Pa[8], audN, MAUD);
  gemm(imgN, KwI, Pi[4],  nullptr, nullptr, imgK, MIMG, DIM, DIM, 0);
  gemm(imgN, VwI, Pi[18], nullptr, nullptr, imgV, MIMG, DIM, DIM, 0);
  gemm(audN, KwA, Pa[4],  nullptr, nullptr, audK, MAUD, DIM, DIM, 0);
  gemm(audN, VwA, Pa[18], nullptr, nullptr, audV, MAUD, DIM, DIM, 0);

  const float scale = 0.044194173824159216f;  // 512^-0.5

  auto run = [&](const float* const* P, const bf16_t* Qw, const bf16_t* Wih,
                 const bf16_t* Whh, const bf16_t* W1, const bf16_t* W2,
                 const bf16_t* ownK, const bf16_t* ownV, int ownN,
                 const bf16_t* crossK, int crossN, float* oSlots, float* oQ,
                 float* oOwn, float* oCross) {
    slots_init<<<MSLOT * DIM / 256, 256, 0, stream>>>(slotsParam, slotsF, prevB);
    for (int it = 0; it < 3; ++it) {
      const bool last = (it == 2);
      ln_rows<<<MSLOT / 8, 256, 0, stream>>>(slotsF, P[11], P[10], lnS, MSLOT);
      gemm(lnS, Qw, P[16], nullptr, oQ, qB, MSLOT, DIM, DIM, 0);
      slot_attention<<<BATCH, 128, 0, stream>>>(
          qB, ownK, ownV, last ? oOwn : nullptr, updB, ownN, scale);
      if (last)
        slot_attention<<<BATCH, 128, 0, stream>>>(
            qB, crossK, nullptr, oCross, nullptr, crossN, scale);
      gemm(updB,  Wih, P[1], nullptr, giF, nullptr, MSLOT, 3 * DIM, DIM, 0);
      gemm(prevB, Whh, P[0], nullptr, ghF, nullptr, MSLOT, 3 * DIM, DIM, 0);
      gru_ew<<<MSLOT * DIM / 256, 256, 0, stream>>>(giF, ghF, slotsF);
      ln_rows<<<MSLOT / 8, 256, 0, stream>>>(slotsF, P[7], P[6], lnS, MSLOT);
      gemm(lnS, W1, P[12], nullptr, nullptr, h1B, MSLOT, HF, DIM, 1);
      gemm(h1B, W2, P[13], slotsF, last ? oSlots : slotsF, prevB, MSLOT, DIM,
           HF, 0);
    }
  };

  run(Pi, QwI, WihI, WhhI, W1I, W2I, imgK, imgV, NI, audK, NA, oImgSlots,
      oImgQ, oII, oIA);
  run(Pa, QwA, WihA, WhhA, W1A, W2A, audK, audV, NA, imgK, NI, oAudSlots,
      oAudQ, oAA, oAI);
}
